// ModLayer_22883585753404
// MI455X (gfx1250) — compile-verified, matrix operand order checked by reading, not run on hardware
//
#include <hip/hip_runtime.h>
#include <hip/hip_bf16.h>

// ---------------------------------------------------------------------------
// Problem constants (from reference)
// ---------------------------------------------------------------------------
#define Dc   2048
#define Hc   16
#define HDc  128
#define FFc  8192
#define Bc   4
#define Sc   2048
#define Kp   1024          // kept tokens per batch row (S * 0.5)
#define Mrows (Bc * Kp)    // 4096 flattened GEMM rows

typedef __bf16 v16bf __attribute__((ext_vector_type(16)));
typedef float  v8f   __attribute__((ext_vector_type(8)));

__device__ __forceinline__ float fast_tanh(float u) {
    return 1.0f - 2.0f / (1.0f + __expf(2.0f * u));
}
__device__ __forceinline__ float gelu_tanh(float x) {
    float x3 = x * x * x;
    return 0.5f * x * (1.0f + fast_tanh(0.7978845608028654f * (x + 0.044715f * x3)));
}

// ---------------------------------------------------------------------------
// CDNA5 async copy: global -> LDS, 16 bytes per lane, tracked by ASYNCcnt.
// VDST = per-lane LDS byte offset (low 32 bits of a generic LDS pointer),
// VADDR = 64-bit global address, no SADDR.
// ---------------------------------------------------------------------------
__device__ __forceinline__ void async_copy16(__bf16* lds, const __bf16* gsrc) {
    unsigned loff = (unsigned)(unsigned long long)lds;
    asm volatile("global_load_async_to_lds_b128 %0, %1, off"
                 :: "v"(loff), "v"(gsrc) : "memory");
}
__device__ __forceinline__ void async_wait0() {
    asm volatile("s_wait_asynccnt 0x0" ::: "memory");
}

// ---------------------------------------------------------------------------
// f32 -> bf16 conversion WITH transpose (weights): src is KxN, dst is NxK.
// ---------------------------------------------------------------------------
__global__ __launch_bounds__(256) void cvt_transpose(const float* __restrict__ src,
                                                     __bf16* __restrict__ dst,
                                                     int Kd, int N) {
    __shared__ float tile[32][33];
    const int k0 = blockIdx.y * 32;
    const int n0 = blockIdx.x * 32;
    const int tx = threadIdx.x & 31;
    const int ty = threadIdx.x >> 5;           // 0..7
#pragma unroll
    for (int i = 0; i < 4; ++i)
        tile[ty + 8 * i][tx] = src[(size_t)(k0 + ty + 8 * i) * N + n0 + tx];
    __syncthreads();
#pragma unroll
    for (int i = 0; i < 4; ++i)
        dst[(size_t)(n0 + ty + 8 * i) * Kd + k0 + tx] = (__bf16)tile[tx][ty + 8 * i];
}

// ---------------------------------------------------------------------------
// Router: logits = h . w, sigmoid, mask.  One wave per token.
// ---------------------------------------------------------------------------
__global__ __launch_bounds__(256) void router_kernel(const float* __restrict__ hs,
                                                     const float* __restrict__ rw,
                                                     const unsigned char* __restrict__ mask,
                                                     float* __restrict__ wout) {
    int token = blockIdx.x * 8 + (threadIdx.x >> 5);
    int lane  = threadIdx.x & 31;
    const float* x = hs + (size_t)token * Dc;
    float s = 0.f;
    for (int i = lane; i < Dc; i += 32) s += x[i] * rw[i];
#pragma unroll
    for (int d = 16; d >= 1; d >>= 1) s += __shfl_xor(s, d, 32);
    if (lane == 0)
        wout[token] = mask[token] ? (1.f / (1.f + __expf(-s))) : 0.f;
}

// ---------------------------------------------------------------------------
// Top-K per batch row: bitonic sort 2048 (w,idx) pairs descending, keep 1024,
// re-sort kept indices ascending, build membership map.
// ---------------------------------------------------------------------------
__global__ __launch_bounds__(1024) void topk_kernel(const float* __restrict__ wts,
                                                    int* __restrict__ idxs,
                                                    int* __restrict__ map) {
    const int b = blockIdx.x;
    const int tid = threadIdx.x;
    __shared__ float sw[Sc];
    __shared__ int   si[Sc];
    __shared__ int   sel[Kp];

    for (int i = tid; i < Sc; i += 1024) { sw[i] = wts[b * Sc + i]; si[i] = i; }

    for (int k2 = 2; k2 <= Sc; k2 <<= 1)
        for (int j = k2 >> 1; j > 0; j >>= 1) {
            __syncthreads();
            for (int i = tid; i < Sc; i += 1024) {
                int x = i ^ j;
                if (x > i) {
                    bool dir = ((i & k2) == 0);
                    float wi = sw[i], wx = sw[x];
                    int   ii = si[i], ix = si[x];
                    bool xFirst = (wx > wi) || (wx == wi && ix < ii);
                    if (xFirst == dir) { sw[i] = wx; sw[x] = wi; si[i] = ix; si[x] = ii; }
                }
            }
        }
    __syncthreads();
    if (tid < Kp) sel[tid] = si[tid];
    for (int k2 = 2; k2 <= Kp; k2 <<= 1)
        for (int j = k2 >> 1; j > 0; j >>= 1) {
            __syncthreads();
            if (tid < Kp) {
                int i = tid, x = i ^ j;
                if (x > i) {
                    bool dir = ((i & k2) == 0);
                    int a = sel[i], c = sel[x];
                    if ((c < a) == dir) { sel[i] = c; sel[x] = a; }
                }
            }
        }
    __syncthreads();
    for (int i = tid; i < Sc; i += 1024) map[b * Sc + i] = -1;
    __syncthreads();
    if (tid < Kp) {
        idxs[b * Kp + tid] = sel[tid];
        map[b * Sc + sel[tid]] = tid;
    }
}

// ---------------------------------------------------------------------------
// Gather kept rows + RMSNorm (ln1). Stores kept f32, h1 f32, gathered pos.
// ---------------------------------------------------------------------------
__global__ __launch_bounds__(256) void gather_norm_kernel(const float* __restrict__ hs,
                                                          const int* __restrict__ idxs,
                                                          const int* __restrict__ pos_ids,
                                                          const float* __restrict__ lnw,
                                                          float* __restrict__ keptf,
                                                          float* __restrict__ h1f,
                                                          int* __restrict__ posk) {
    const int row = blockIdx.x;               // b*Kp + k
    const int b   = row / Kp;
    const int s   = idxs[row];
    const float* src = hs + ((size_t)b * Sc + s) * Dc;
    const int tid = threadIdx.x;
    __shared__ float red[256];

    float v[8], ss = 0.f;
#pragma unroll
    for (int i = 0; i < 8; ++i) { v[i] = src[tid + i * 256]; ss += v[i] * v[i]; }
    red[tid] = ss; __syncthreads();
    for (int s2 = 128; s2 > 0; s2 >>= 1) { if (tid < s2) red[tid] += red[tid + s2]; __syncthreads(); }
    float rn = rsqrtf(red[0] / (float)Dc + 1e-6f);

    float* kdst = keptf + (size_t)row * Dc;
    float* hdst = h1f   + (size_t)row * Dc;
#pragma unroll
    for (int i = 0; i < 8; ++i) {
        int d = tid + i * 256;
        kdst[d] = v[i];
        hdst[d] = v[i] * rn * lnw[d];
    }
    if (tid == 0) posk[row] = pos_ids[s];
}

// ---------------------------------------------------------------------------
// Residual + RMSNorm (ln2): h2n = rmsnorm(kept + ao, ln2)
// ---------------------------------------------------------------------------
__global__ __launch_bounds__(256) void norm2_kernel(const float* __restrict__ keptf,
                                                    const float* __restrict__ aof,
                                                    const float* __restrict__ lnw,
                                                    float* __restrict__ h2n) {
    const int row = blockIdx.x;
    const int tid = threadIdx.x;
    __shared__ float red[256];
    const float* a = keptf + (size_t)row * Dc;
    const float* o = aof   + (size_t)row * Dc;
    float v[8], ss = 0.f;
#pragma unroll
    for (int i = 0; i < 8; ++i) {
        int d = tid + i * 256;
        v[i] = a[d] + o[d]; ss += v[i] * v[i];
    }
    red[tid] = ss; __syncthreads();
    for (int s2 = 128; s2 > 0; s2 >>= 1) { if (tid < s2) red[tid] += red[tid + s2]; __syncthreads(); }
    float rn = rsqrtf(red[0] / (float)Dc + 1e-6f);
    float* dst = h2n + (size_t)row * Dc;
#pragma unroll
    for (int i = 0; i < 8; ++i) { int d = tid + i * 256; dst[d] = v[i] * rn * lnw[d]; }
}

// ---------------------------------------------------------------------------
// Tiled WMMA GEMM:  C[M,N] = A[M,Kd] (f32 or bf16) x W (bf16, TRANSPOSED NxKd)
// Block tile 128x128x32, 256 threads = 8 waves (4M x 2N), DOUBLE-BUFFERED LDS.
// B (and bf16 A) staged with global_load_async_to_lds_b128 (ASYNCcnt).
// EPI: 0 = f32 store, 2 = RoPE -> bf16, 3 = GELU -> bf16,
//      1 = bf16 store transposed per head ([b][h][dim][token]) for V.
// ---------------------------------------------------------------------------
#define BM 128
#define BN 128
#define BKt 32
#define LPAD 8

template <int EPI, bool ABF16>
__global__ __launch_bounds__(256) void gemm_wmma(const void* __restrict__ Ap,
                                                 const __bf16* __restrict__ Wt,
                                                 void* __restrict__ outp,
                                                 int Kd, int N,
                                                 const int* __restrict__ posk) {
    __shared__ __bf16 As[2][BM][BKt + LPAD];      // [m][k]
    __shared__ __bf16 Bs[2][BN][BKt + LPAD];      // [n][k]  (weights pre-transposed)

    const int tid  = threadIdx.x;
    const int lane = tid & 31;
    const int wave = tid >> 5;
    const int wm = wave >> 1, wn = wave & 1;
    const int lm = lane & 15, lh = lane >> 4;
    const int m0 = blockIdx.y * BM;
    const int n0 = blockIdx.x * BN;

    v8f acc[2][4];
#pragma unroll
    for (int i = 0; i < 2; ++i)
#pragma unroll
        for (int j = 0; j < 4; ++j) acc[i][j] = (v8f){0.f,0.f,0.f,0.f,0.f,0.f,0.f,0.f};

    const int srow = tid >> 1, scol = (tid & 1) * 16;   // 128 x 32 staging
    const int kTiles = Kd / BKt;

    auto stage = [&](int kt, int buf) {
        const int k0 = kt * BKt;
        if (ABF16) {
            const __bf16* src = (const __bf16*)Ap + (size_t)(m0 + srow) * Kd + k0 + scol;
            async_copy16(&As[buf][srow][scol],     src);
            async_copy16(&As[buf][srow][scol + 8], src + 8);
        } else {
            const float* src = (const float*)Ap + (size_t)(m0 + srow) * Kd + k0 + scol;
#pragma unroll
            for (int i = 0; i < 16; ++i) As[buf][srow][scol + i] = (__bf16)src[i];
        }
        const __bf16* wsrc = Wt + (size_t)(n0 + srow) * Kd + k0 + scol;
        async_copy16(&Bs[buf][srow][scol],     wsrc);
        async_copy16(&Bs[buf][srow][scol + 8], wsrc + 8);
    };

    stage(0, 0);
    for (int kt = 0; kt < kTiles; ++kt) {
        const int cur = kt & 1;
        async_wait0();
        __syncthreads();                     // buffer `cur` ready; prev compute done
        if (kt + 1 < kTiles) stage(kt + 1, cur ^ 1);

        v16bf afr[2];
#pragma unroll
        for (int fm = 0; fm < 2; ++fm) {
            const int r = wm * 32 + fm * 16 + lm;
#pragma unroll
            for (int j = 0; j < 8; ++j) {
                afr[fm][j]     = As[cur][r][lh * 8 + j];
                afr[fm][8 + j] = As[cur][r][16 + lh * 8 + j];
            }
        }
#pragma unroll
        for (int fn = 0; fn < 4; ++fn) {
            const int c = fn * 32 + wn * 16 + lm;
            v16bf bfr;
#pragma unroll
            for (int j = 0; j < 8; ++j) {
                bfr[j]     = Bs[cur][c][lh * 8 + j];
                bfr[8 + j] = Bs[cur][c][16 + lh * 8 + j];
            }
#pragma unroll
            for (int fm = 0; fm < 2; ++fm)
                acc[fm][fn] = __builtin_amdgcn_wmma_f32_16x16x32_bf16(
                    false, afr[fm], false, bfr, (short)0, acc[fm][fn], false, false);
        }
    }

    // ---- epilogue ----
    if constexpr (EPI == 0) {
        float* Co = (float*)outp;
#pragma unroll
        for (int fm = 0; fm < 2; ++fm)
#pragma unroll
            for (int fn = 0; fn < 4; ++fn) {
                const int col = n0 + fn * 32 + wn * 16 + lm;
#pragma unroll
                for (int r = 0; r < 8; ++r) {
                    const int row = m0 + wm * 32 + fm * 16 + r + 8 * lh;
                    Co[(size_t)row * N + col] = acc[fm][fn][r];
                }
            }
    } else if constexpr (EPI == 3) {
        __bf16* Co = (__bf16*)outp;
#pragma unroll
        for (int fm = 0; fm < 2; ++fm)
#pragma unroll
            for (int fn = 0; fn < 4; ++fn) {
                const int col = n0 + fn * 32 + wn * 16 + lm;
#pragma unroll
                for (int r = 0; r < 8; ++r) {
                    const int row = m0 + wm * 32 + fm * 16 + r + 8 * lh;
                    Co[(size_t)row * N + col] = (__bf16)gelu_tanh(acc[fm][fn][r]);
                }
            }
    } else if constexpr (EPI == 1) {
        // V: store transposed per head: vbT[((b*H + h)*HD + dim)*Kp + token]
        __bf16* Co = (__bf16*)outp;
        const int hIdx = n0 / HDc;
#pragma unroll
        for (int fm = 0; fm < 2; ++fm) {
            const int rowBase = m0 + wm * 32 + fm * 16 + 8 * lh;
            const int bIdx = rowBase / Kp;
            const int tok0 = rowBase - bIdx * Kp;
#pragma unroll
            for (int fn = 0; fn < 4; ++fn) {
                const int dim = fn * 32 + wn * 16 + lm;
                __bf16* dst = Co + (((size_t)(bIdx * Hc + hIdx) * HDc + dim) * Kp) + tok0;
#pragma unroll
                for (int r = 0; r < 8; ++r) dst[r] = (__bf16)acc[fm][fn][r];
            }
        }
    } else {  // EPI == 2: RoPE. n0 is a head base (BN == HD == 128).
        __bf16* Co = (__bf16*)outp;
#pragma unroll
        for (int fm = 0; fm < 2; ++fm) {
            const int rowBase = m0 + wm * 32 + fm * 16 + 8 * lh;
            float pos[8];
#pragma unroll
            for (int r = 0; r < 8; ++r) pos[r] = (float)posk[rowBase + r];
#pragma unroll
            for (int fn = 0; fn < 2; ++fn) {
                const int hc = fn * 32 + wn * 16 + lm;                  // 0..63
                const float freq = __expf(-(float)hc * (9.210340371976184f / 64.f));
#pragma unroll
                for (int r = 0; r < 8; ++r) {
                    float sn, cs;
                    __sincosf(pos[r] * freq, &sn, &cs);
                    float x1 = acc[fm][fn][r];
                    float x2 = acc[fm][fn + 2][r];
                    const size_t o = (size_t)(rowBase + r) * N + n0 + hc;
                    Co[o]      = (__bf16)(x1 * cs - x2 * sn);
                    Co[o + 64] = (__bf16)(x2 * cs + x1 * sn);
                }
            }
        }
    }
}

// ---------------------------------------------------------------------------
// Flash-style causal attention over the compressed sequence.
// Grid: (Kp/64, H, B); 128 threads = 4 waves, each wave owns 16 queries.
// Q/K in [token][D] bf16; V pre-transposed [b][h][dim][token] bf16.
// K/V chunks staged via async-to-LDS; scores and P.V via WMMA.
// ---------------------------------------------------------------------------
__global__ __launch_bounds__(128) void attn_kernel(const __bf16* __restrict__ Q,
                                                   const __bf16* __restrict__ Kb,
                                                   const __bf16* __restrict__ Vt,
                                                   float* __restrict__ O) {
    const int b = blockIdx.z, h = blockIdx.y;
    const int qblk = blockIdx.x * 64;
    const int tid = threadIdx.x;
    const int wave = tid >> 5, lane = tid & 31;
    const int lm = lane & 15, lh = lane >> 4;
    const int qbase = qblk + wave * 16;
    const size_t headOff = (size_t)h * HDc;

    __shared__ __bf16 Ks[32][HDc + LPAD];      // [key][dim]
    __shared__ __bf16 Vs[HDc][32 + LPAD];      // [dim][key]  (transposed)
    __shared__ __bf16 Ps[4][16][32 + LPAD];

    // preload the wave's 16x128 Q tile as 4 A-fragments
    v16bf qa[4];
    {
        const __bf16* qp = Q + ((size_t)(b * Kp + qbase + lm)) * Dc + headOff;
#pragma unroll
        for (int ks = 0; ks < 4; ++ks)
#pragma unroll
            for (int j = 0; j < 8; ++j) {
                qa[ks][j]     = qp[ks * 32 + lh * 8 + j];
                qa[ks][8 + j] = qp[ks * 32 + 16 + lh * 8 + j];
            }
    }

    v8f ob[8];
#pragma unroll
    for (int i = 0; i < 8; ++i) ob[i] = (v8f){0.f,0.f,0.f,0.f,0.f,0.f,0.f,0.f};
    float mrun[8], lrun[8];
#pragma unroll
    for (int r = 0; r < 8; ++r) { mrun[r] = -1e30f; lrun[r] = 0.f; }

    const int nch = qblk / 32 + 2;
    const int skey = tid >> 2, sdim = (tid & 3) * 32;   // K staging: 32x128
    const __bf16* vhead = Vt + ((size_t)(b * Hc + h) * HDc) * Kp;

    for (int kc = 0; kc < nch; ++kc) {
        const int kbk = kc * 32;
        __syncthreads();                      // previous chunk fully consumed
        {
            const __bf16* kp = Kb + ((size_t)(b * Kp + kbk + skey)) * Dc + headOff + sdim;
#pragma unroll
            for (int i = 0; i < 4; ++i) async_copy16(&Ks[skey][sdim + 8 * i], kp + 8 * i);
            const __bf16* vp = vhead + (size_t)tid * Kp + kbk;   // tid = dim 0..127
#pragma unroll
            for (int i = 0; i < 4; ++i) async_copy16(&Vs[tid][8 * i], vp + 8 * i);
        }
        async_wait0();
        __syncthreads();

        if (kbk <= qbase + 15) {
            float pt[2][8];
#pragma unroll
            for (int sub = 0; sub < 2; ++sub) {
                v8f s = (v8f){0.f,0.f,0.f,0.f,0.f,0.f,0.f,0.f};
                const int key = sub * 16 + lm;
#pragma unroll
                for (int ks = 0; ks < 4; ++ks) {
                    v16bf kfr;
#pragma unroll
                    for (int j = 0; j < 8; ++j) {
                        kfr[j]     = Ks[key][ks * 32 + lh * 8 + j];
                        kfr[8 + j] = Ks[key][ks * 32 + 16 + lh * 8 + j];
                    }
                    s = __builtin_amdgcn_wmma_f32_16x16x32_bf16(
                        false, qa[ks], false, kfr, (short)0, s, false, false);
                }
#pragma unroll
                for (int r = 0; r < 8; ++r) {
                    const int qi = qbase + r + 8 * lh;
                    const int ki = kbk + sub * 16 + lm;
                    float v = s[r] * 0.08838834764831845f;   // 1/sqrt(128)
                    pt[sub][r] = (ki > qi) ? -1e30f : v;
                }
            }
            // online softmax per row (rows live across the 16-lane halves)
#pragma unroll
            for (int r = 0; r < 8; ++r) {
                float mx = fmaxf(pt[0][r], pt[1][r]);
#pragma unroll
                for (int d = 8; d >= 1; d >>= 1) mx = fmaxf(mx, __shfl_xor(mx, d, 16));
                const float nm = fmaxf(mrun[r], mx);
                const float sc = __expf(mrun[r] - nm);
                mrun[r] = nm;
                float p0 = __expf(pt[0][r] - nm);
                float p1 = __expf(pt[1][r] - nm);
                float rs = p0 + p1;
#pragma unroll
                for (int d = 8; d >= 1; d >>= 1) rs += __shfl_xor(rs, d, 16);
                lrun[r] = lrun[r] * sc + rs;
#pragma unroll
                for (int fn = 0; fn < 8; ++fn) ob[fn][r] = ob[fn][r] * sc;
                Ps[wave][r + 8 * lh][lm]      = (__bf16)p0;
                Ps[wave][r + 8 * lh][16 + lm] = (__bf16)p1;
            }
            // P (16x32) as A-fragment from per-wave LDS tile (intra-wave, DS in-order)
            v16bf pa;
#pragma unroll
            for (int j = 0; j < 8; ++j) {
                pa[j]     = Ps[wave][lm][lh * 8 + j];
                pa[8 + j] = Ps[wave][lm][16 + lh * 8 + j];
            }
#pragma unroll
            for (int fn = 0; fn < 8; ++fn) {
                const int dim = fn * 16 + lm;
                v16bf vfr;
#pragma unroll
                for (int j = 0; j < 8; ++j) {
                    vfr[j]     = Vs[dim][lh * 8 + j];
                    vfr[8 + j] = Vs[dim][16 + lh * 8 + j];
                }
                ob[fn] = __builtin_amdgcn_wmma_f32_16x16x32_bf16(
                    false, pa, false, vfr, (short)0, ob[fn], false, false);
            }
        }
    }

#pragma unroll
    for (int r = 0; r < 8; ++r) {
        const int row = qbase + r + 8 * lh;
        const float inv = 1.f / lrun[r];
#pragma unroll
        for (int fn = 0; fn < 8; ++fn)
            O[((size_t)(b * Kp + row)) * Dc + headOff + fn * 16 + lm] = ob[fn][r] * inv;
    }
}

// ---------------------------------------------------------------------------
// Finalize: out = hidden everywhere; selected rows get delta; scale by router
// weight where mask; add kept (== hidden row) back at selected rows.
// ---------------------------------------------------------------------------
__global__ __launch_bounds__(256) void finalize_kernel(const float* __restrict__ hs,
                                                       const float* __restrict__ wts,
                                                       const unsigned char* __restrict__ mask,
                                                       const int* __restrict__ map,
                                                       const float* __restrict__ aoproj,
                                                       const float* __restrict__ mlpf,
                                                       float* __restrict__ out) {
    const int t = blockIdx.x;            // b*S + s
    const int b = t / Sc;
    const float w = wts[t];
    const bool  m = mask[t] != 0;
    const int   kk = map[t];
    const float* hrow = hs + (size_t)t * Dc;
    float* orow = out + (size_t)t * Dc;
    for (int i = threadIdx.x; i < Dc; i += 256) {
        const float h = hrow[i];
        float o;
        if (kk >= 0) {
            const size_t off = ((size_t)(b * Kp + kk)) * Dc + i;
            const float delta = aoproj[off] + mlpf[off];
            o = m ? delta * w + h : delta + h;
        } else {
            o = m ? h * w : h;
        }
        orow[i] = o;
    }
}

// ---------------------------------------------------------------------------
// Launch
// ---------------------------------------------------------------------------
extern "C" void kernel_launch(void* const* d_in, const int* in_sizes, int n_in,
                              void* d_out, int out_size, void* d_ws, size_t ws_size,
                              hipStream_t stream) {
    const float* hs   = (const float*)d_in[0];
    const int*   pos  = (const int*)d_in[1];
    const unsigned char* mmask = (const unsigned char*)d_in[2];
    // d_in[3] attention_mask_2d: unused by reference
    const float* rw  = (const float*)d_in[4];
    const float* ln1 = (const float*)d_in[5];
    const float* ln2 = (const float*)d_in[6];
    const float* Wq  = (const float*)d_in[7];
    const float* Wk  = (const float*)d_in[8];
    const float* Wv  = (const float*)d_in[9];
    const float* Wo  = (const float*)d_in[10];
    const float* W1  = (const float*)d_in[11];
    const float* W2  = (const float*)d_in[12];
    float* out = (float*)d_out;

    char* ws = (char*)d_ws;
    size_t off = 0;
    auto carve = [&](size_t bytes) -> char* {
        char* p = ws + off;
        off += (bytes + 255) & ~(size_t)255;
        return p;
    };

    // transposed bf16 weights
    __bf16* wq_t = (__bf16*)carve((size_t)Dc * Dc * 2);
    __bf16* wk_t = (__bf16*)carve((size_t)Dc * Dc * 2);
    __bf16* wv_t = (__bf16*)carve((size_t)Dc * Dc * 2);
    __bf16* wo_t = (__bf16*)carve((size_t)Dc * Dc * 2);
    __bf16* w1_t = (__bf16*)carve((size_t)Dc * FFc * 2);
    __bf16* w2_t = (__bf16*)carve((size_t)FFc * Dc * 2);
    float*  wtok = (float*)carve((size_t)Bc * Sc * 4);
    int*    idxs = (int*)carve((size_t)Bc * Kp * 4);
    int*    map  = (int*)carve((size_t)Bc * Sc * 4);
    int*    posk = (int*)carve((size_t)Bc * Kp * 4);
    float*  keptf = (float*)carve((size_t)Mrows * Dc * 4);
    float*  h1f   = (float*)carve((size_t)Mrows * Dc * 4);
    __bf16* qb  = (__bf16*)carve((size_t)Mrows * Dc * 2);
    __bf16* kbb = (__bf16*)carve((size_t)Mrows * Dc * 2);
    __bf16* vbT = (__bf16*)carve((size_t)Mrows * Dc * 2);  // [b][h][dim][token]
    float*  attnf   = (float*)carve((size_t)Mrows * Dc * 4);
    float*  aoprojf = (float*)carve((size_t)Mrows * Dc * 4);
    float*  h2n     = (float*)carve((size_t)Mrows * Dc * 4);
    __bf16* midb    = (__bf16*)carve((size_t)Mrows * FFc * 2);
    float*  mlpf    = (float*)carve((size_t)Mrows * Dc * 4);
    (void)ws_size; (void)in_sizes; (void)n_in; (void)out_size;

    // 1) weight conversion + transpose  (src KxN -> dst NxK bf16)
    cvt_transpose<<<dim3(Dc / 32, Dc / 32), 256, 0, stream>>>(Wq, wq_t, Dc, Dc);
    cvt_transpose<<<dim3(Dc / 32, Dc / 32), 256, 0, stream>>>(Wk, wk_t, Dc, Dc);
    cvt_transpose<<<dim3(Dc / 32, Dc / 32), 256, 0, stream>>>(Wv, wv_t, Dc, Dc);
    cvt_transpose<<<dim3(Dc / 32, Dc / 32), 256, 0, stream>>>(Wo, wo_t, Dc, Dc);
    cvt_transpose<<<dim3(FFc / 32, Dc / 32), 256, 0, stream>>>(W1, w1_t, Dc, FFc);
    cvt_transpose<<<dim3(Dc / 32, FFc / 32), 256, 0, stream>>>(W2, w2_t, FFc, Dc);

    // 2) router + top-k
    router_kernel<<<(Bc * Sc) / 8, 256, 0, stream>>>(hs, rw, mmask, wtok);
    topk_kernel<<<Bc, 1024, 0, stream>>>(wtok, idxs, map);

    // 3) gather + rmsnorm(ln1)
    gather_norm_kernel<<<Mrows, 256, 0, stream>>>(hs, idxs, pos, ln1, keptf, h1f, posk);

    // 4) Q/K/V projections (WMMA); RoPE fused for Q,K; V stored head-transposed
    dim3 gD(Dc / BN, Mrows / BM);
    gemm_wmma<2, false><<<gD, 256, 0, stream>>>(h1f, wq_t, qb,  Dc, Dc, posk);
    gemm_wmma<2, false><<<gD, 256, 0, stream>>>(h1f, wk_t, kbb, Dc, Dc, posk);
    gemm_wmma<1, false><<<gD, 256, 0, stream>>>(h1f, wv_t, vbT, Dc, Dc, nullptr);

    // 5) causal flash attention (WMMA)
    attn_kernel<<<dim3(Kp / 64, Hc, Bc), 128, 0, stream>>>(qb, kbb, vbT, attnf);

    // 6) O-projection
    gemm_wmma<0, false><<<gD, 256, 0, stream>>>(attnf, wo_t, aoprojf, Dc, Dc, nullptr);

    // 7) residual + rmsnorm(ln2)
    norm2_kernel<<<Mrows, 256, 0, stream>>>(keptf, aoprojf, ln2, h2n);

    // 8) MLP: W1 (GELU epilogue, bf16 mid) then W2
    dim3 gFF(FFc / BN, Mrows / BM);
    gemm_wmma<3, false><<<gFF, 256, 0, stream>>>(h2n, w1_t, midb, Dc, FFc, nullptr);
    gemm_wmma<0, true><<<gD, 256, 0, stream>>>(midb, w2_t, mlpf, FFc, Dc, nullptr);

    // 9) scatter/scale/residual
    finalize_kernel<<<Bc * Sc, 256, 0, stream>>>(hs, wtok, mmask, map, aoprojf, mlpf, out);
}
